// OscillatoryCycle_3255585210482
// MI455X (gfx1250) — matrix operand order repaired and checked by reading, not compiled, not run on hardware
//
#include <hip/hip_runtime.h>
#include <math.h>

// ---------------------------------------------------------------- types
typedef __bf16 bf16;
typedef bf16  v16bf __attribute__((ext_vector_type(16)));
typedef bf16  v8bf  __attribute__((ext_vector_type(8)));
typedef float v8f   __attribute__((ext_vector_type(8)));

#define BB 4
#define TT 2048
#define VV 1024
#define CC 256

// B operands live in "KS32" layout: element (k,n) of a K x N matrix at
//   ((k/32)*N + n)*32 + (k%32)
// so a 32(K) x TN(N) LDS tile is a contiguous copy, and the WMMA B fragment
// (lane = N column, k 8h..8h+7 and 16+8h..16+8h+7 for lane half h) is two 16B reads of one LDS row.

// ---------------------------------------------------------------- band weights
// W[v][c] = sum_j basis[v][j] * coef[c][j]; optional softmax over v (axis 0).
// transposed==0: KS32 with K=v (N=CC).  transposed==1: KS32 with K=c (N=VV).
__global__ __launch_bounds__(256)
void band_weights_kernel(const float* __restrict__ basis,
                         const float* __restrict__ coef,
                         int K2, bf16* __restrict__ out,
                         int softmax, int transposed) {
  __shared__ float cf[128];
  __shared__ float red[256];
  const int c = blockIdx.x;
  const int tid = threadIdx.x;
  if (tid < K2) cf[tid] = coef[(size_t)c * K2 + tid];
  __syncthreads();
  float vals[4];
#pragma unroll
  for (int i = 0; i < 4; ++i) {
    const int v = tid + i * 256;
    const float* brow = basis + (size_t)v * K2;
    float acc = 0.f;
    for (int j = 0; j < K2; ++j) acc += brow[j] * cf[j];
    vals[i] = acc;
  }
  if (softmax) {
    float m = vals[0];
#pragma unroll
    for (int i = 1; i < 4; ++i) m = fmaxf(m, vals[i]);
    red[tid] = m; __syncthreads();
    for (int s = 128; s > 0; s >>= 1) {
      if (tid < s) red[tid] = fmaxf(red[tid], red[tid + s]);
      __syncthreads();
    }
    const float gmax = red[0]; __syncthreads();
    float lsum = 0.f;
#pragma unroll
    for (int i = 0; i < 4; ++i) { vals[i] = __expf(vals[i] - gmax); lsum += vals[i]; }
    red[tid] = lsum; __syncthreads();
    for (int s = 128; s > 0; s >>= 1) {
      if (tid < s) red[tid] += red[tid + s];
      __syncthreads();
    }
    const float inv = 1.f / red[0];
#pragma unroll
    for (int i = 0; i < 4; ++i) vals[i] *= inv;
  }
#pragma unroll
  for (int i = 0; i < 4; ++i) {
    const int v = tid + i * 256;
    if (transposed) out[(((size_t)(c >> 5) * VV + v) << 5) + (c & 31)] = (bf16)vals[i];
    else            out[(((size_t)(v >> 5) * CC + c) << 5) + (v & 31)] = (bf16)vals[i];
  }
}

// ---------------------------------------------------------------- rmsnorm -> bf16 (row-major)
__global__ __launch_bounds__(256)
void rmsnorm_bf16_kernel(const float* __restrict__ x, bf16* __restrict__ out, int rowlen) {
  __shared__ float red[256];
  const int row = blockIdx.x;
  const int tid = threadIdx.x;
  const float* p = x + (size_t)row * rowlen;
  float s = 0.f;
  for (int i = tid; i < rowlen; i += 256) { const float v = p[i]; s += v * v; }
  red[tid] = s; __syncthreads();
  for (int t = 128; t > 0; t >>= 1) {
    if (tid < t) red[tid] += red[tid + t];
    __syncthreads();
  }
  const float scale = rsqrtf(red[0] / (float)rowlen + 1.1920929e-07f);
  bf16* o = out + (size_t)row * rowlen;
  for (int i = tid; i < rowlen; i += 256) o[i] = (bf16)(p[i] * scale);
}

// ---------------------------------------------------------------- WMMA GEMM
// C[M][N](f32, optional) = A[M][K](bf16 row-major) x B[K][N](bf16 KS32)
// Optional bf16 output D: dmode 1=row-major, 2=KS32 over rows (v layout,N=CC),
// 3=KS32 over cols (k-transposed layout, N=TT per batch).
#define TM 256
#define TN 64
#define TK 32
#define LP 40   // padded LDS row length: 80B stride -> 16B aligned, conflict-free

__device__ __forceinline__ void store_d(bf16* D, int dmode, int r, int col, int N, float v) {
  if (dmode == 1) {
    D[(size_t)r * N + col] = (bf16)v;
  } else if (dmode == 2) {           // per-batch KS32, K=T rows, N=CC
    const int b = r >> 11, t = r & (TT - 1);
    D[(size_t)b * TT * CC + (((size_t)(t >> 5) * CC + col) << 5) + (t & 31)] = (bf16)v;
  } else {                           // per-batch KS32 of transpose, K=col(C), N=TT
    const int b = r >> 11, t = r & (TT - 1);
    D[(size_t)b * CC * TT + (((size_t)(col >> 5) * TT + t) << 5) + (col & 31)] = (bf16)v;
  }
}

__global__ __launch_bounds__(256)
void gemm_bf16_kernel(const bf16* __restrict__ A, const bf16* __restrict__ B,
                      float* __restrict__ C, bf16* __restrict__ D, int dmode,
                      int M, int N, int K) {
  __shared__ __align__(64) bf16 As[2][TM][LP];   // 40 KB
  __shared__ __align__(64) bf16 Bs[2][TN][LP];   // 10 KB
  const int m0 = blockIdx.y * TM;
  const int n0 = blockIdx.x * TN;
  const int tid  = threadIdx.x;
  const int lane = tid & 31;
  const int wave = tid >> 5;
  const int hsel = (lane < 16) ? 0 : 1;
  const int lrow = lane & 15;

  const bf16* Arow = A + (size_t)(m0 + tid) * K;          // thread stages A row `tid`
  const int   bn   = tid >> 2;                            // B stage: n within tile
  const int   bk   = (tid & 3) << 3;                      // B stage: k offset
  const size_t bsrc0 = (size_t)n0 * 32 + (size_t)tid * 8; // within a k-block slice

  v8f acc0[4], acc1[4];
#pragma unroll
  for (int j = 0; j < 4; ++j)
#pragma unroll
    for (int e = 0; e < 8; ++e) { acc0[j][e] = 0.f; acc1[j][e] = 0.f; }

  // ---- stage k-block 0 into buffer 0
  {
    const bf16* as = Arow;
    bf16* ad = &As[0][tid][0];
    *(v8bf*)(ad)      = *(const v8bf*)(as);
    *(v8bf*)(ad + 8)  = *(const v8bf*)(as + 8);
    *(v8bf*)(ad + 16) = *(const v8bf*)(as + 16);
    *(v8bf*)(ad + 24) = *(const v8bf*)(as + 24);
    *(v8bf*)(&Bs[0][bn][bk]) = *(const v8bf*)(B + bsrc0);
  }
  __syncthreads();

  int cur = 0;
  for (int k0 = 0; k0 < K; k0 += TK) {
    const bool pre = (k0 + TK) < K;
    v8bf pa0, pa1, pa2, pa3, pb;
    if (pre) {  // prefetch next k-block into registers (waits deferred past WMMAs)
      const bf16* as = Arow + k0 + TK;
      pa0 = *(const v8bf*)(as);
      pa1 = *(const v8bf*)(as + 8);
      pa2 = *(const v8bf*)(as + 16);
      pa3 = *(const v8bf*)(as + 24);
      pb  = *(const v8bf*)(B + (size_t)((k0 >> 5) + 1) * N * 32 + bsrc0);
    }

    // A fragments for the two M-subtiles this wave owns
    union { v16bf v; v8bf h[2]; } af0, af1;
    {
      const bf16* ar0 = &As[cur][wave * 16 + lrow][0];
      const bf16* ar1 = &As[cur][128 + wave * 16 + lrow][0];
      af0.h[0] = *(const v8bf*)(ar0 + hsel * 8);
      af0.h[1] = *(const v8bf*)(ar0 + 16 + hsel * 8);
      af1.h[0] = *(const v8bf*)(ar1 + hsel * 8);
      af1.h[1] = *(const v8bf*)(ar1 + 16 + hsel * 8);
    }
#pragma unroll
    for (int j = 0; j < 4; ++j) {
      union { v16bf v; v8bf h[2]; } bu;
      const bf16* br = &Bs[cur][j * 16 + lrow][0];
      bu.h[0] = *(const v8bf*)(br + hsel * 8);
      bu.h[1] = *(const v8bf*)(br + 16 + hsel * 8);
      acc0[j] = __builtin_amdgcn_wmma_f32_16x16x32_bf16(
          false, af0.v, false, bu.v, (short)0, acc0[j], false, false);
      acc1[j] = __builtin_amdgcn_wmma_f32_16x16x32_bf16(
          false, af1.v, false, bu.v, (short)0, acc1[j], false, false);
    }
    __syncthreads();
    if (pre) {
      bf16* ad = &As[cur ^ 1][tid][0];
      *(v8bf*)(ad)      = pa0;
      *(v8bf*)(ad + 8)  = pa1;
      *(v8bf*)(ad + 16) = pa2;
      *(v8bf*)(ad + 24) = pa3;
      *(v8bf*)(&Bs[cur ^ 1][bn][bk]) = pb;
    }
    __syncthreads();
    cur ^= 1;
  }

  // ---- epilogue: D layout VGPR i -> M=i (+8 upper lanes), N = lane%16
#pragma unroll
  for (int j = 0; j < 4; ++j)
#pragma unroll
    for (int i = 0; i < 8; ++i) {
      const int r0 = m0 + wave * 16 + i + (hsel ? 8 : 0);
      const int r1 = r0 + 128;
      const int col = n0 + j * 16 + lrow;
      if (C) {
        C[(size_t)r0 * N + col] = acc0[j][i];
        C[(size_t)r1 * N + col] = acc1[j][i];
      }
      if (D) {
        store_d(D, dmode, r0, col, N, acc0[j][i]);
        store_d(D, dmode, r1, col, N, acc1[j][i]);
      }
    }
}

// ---------------------------------------------------------------- elementwise
__global__ __launch_bounds__(256)
void conv_bf16_kernel(const float* __restrict__ src, bf16* __restrict__ dst, int n) {
  const int i = blockIdx.x * 256 + threadIdx.x;
  if (i < n) dst[i] = (bf16)src[i];
}

// f32 row-major [batch][K][N] (or transposed-source) -> bf16 KS32 per batch
__global__ __launch_bounds__(256)
void conv_ks32_kernel(const float* __restrict__ src, bf16* __restrict__ dst,
                      int K, int N, int batch, int tsrc) {
  const int i = blockIdx.x * 256 + threadIdx.x;
  if (i < batch * K * N) {
    const int b = i / (K * N);
    const int r = i - b * (K * N);
    const int k = r / N;
    const int n = r - k * N;
    const float v = tsrc ? src[(size_t)b * N * K + (size_t)n * K + k]
                         : src[(size_t)b * K * N + (size_t)k * N + n];
    dst[(size_t)b * K * N + (((size_t)(k >> 5) * N + n) << 5) + (k & 31)] = (bf16)v;
  }
}

// masked scores -> bf16 row-major: out[t][s] = (s>t) ? in * sigmoid(dl)^(s-t-1) : 0
__global__ __launch_bounds__(256)
void decay_mask_kernel(const float* __restrict__ scores, bf16* __restrict__ out,
                       const float* __restrict__ decay_logit) {
  const int i = blockIdx.x * 256 + threadIdx.x;
  if (i < TT * TT) {
    const int t = i / TT;
    const int s = i - t * TT;
    const int diff = s - t;
    float v = 0.f;
    if (diff > 0) {
      const float d = 1.f / (1.f + __expf(-decay_logit[0]));
      v = scores[i] * __powf(d, (float)(diff - 1));
    }
    out[i] = (bf16)v;
  }
}

// target *= sigmoid(pre + bias[col])
__global__ __launch_bounds__(256)
void gate_mul_kernel(float* __restrict__ target, const float* __restrict__ pre,
                     const float* __restrict__ bias, int n) {
  const int i = blockIdx.x * 256 + threadIdx.x;
  if (i < n) {
    const float g = 1.f / (1.f + __expf(-(pre[i] + bias[i % CC])));
    target[i] *= g;
  }
}

// h = gelu_exact(h + bias[col])
__global__ __launch_bounds__(256)
void bias_gelu_kernel(float* __restrict__ h, const float* __restrict__ bias, int n) {
  const int i = blockIdx.x * 256 + threadIdx.x;
  if (i < n) {
    const float v = h[i] + bias[i % CC];
    h[i] = 0.5f * v * (1.f + erff(v * 0.70710678118654752f));
  }
}

// dst += src * (*s1) * (*s2)
__global__ __launch_bounds__(256)
void axpy2_kernel(float* __restrict__ dst, const float* __restrict__ src,
                  const float* __restrict__ s1, const float* __restrict__ s2, int n) {
  const int i = blockIdx.x * 256 + threadIdx.x;
  if (i < n) dst[i] += src[i] * s1[0] * s2[0];
}

// ---------------------------------------------------------------- host side
static inline dim3 gemm_grid(int M, int N) { return dim3(N / TN, M / TM); }
static inline int  ceil_blocks(int n) { return (n + 255) / 256; }

extern "C" void kernel_launch(void* const* d_in, const int* in_sizes, int n_in,
                              void* d_out, int out_size, void* d_ws, size_t ws_size,
                              hipStream_t stream) {
  (void)in_sizes; (void)n_in; (void)out_size; (void)ws_size;
  const float* x          = (const float*)d_in[0];
  const float* basis_low  = (const float*)d_in[1];
  const float* basis_mid  = (const float*)d_in[2];
  const float* basis_high = (const float*)d_in[3];
  const float* slow_q = (const float*)d_in[4];
  const float* slow_k = (const float*)d_in[5];
  const float* slow_v = (const float*)d_in[6];
  const float* slow_o = (const float*)d_in[7];
  const float* slow_decay = (const float*)d_in[8];
  const float* slow_scale = (const float*)d_in[9];
  const float* fast_q = (const float*)d_in[10];
  const float* fast_k = (const float*)d_in[11];
  const float* fast_v = (const float*)d_in[12];
  const float* fast_o = (const float*)d_in[13];
  const float* fast_decay = (const float*)d_in[14];
  const float* fast_scale = (const float*)d_in[15];
  const float* low_read  = (const float*)d_in[16];
  const float* low_write = (const float*)d_in[17];
  const float* low_mix   = (const float*)d_in[18];
  const float* low_bias  = (const float*)d_in[19];
  const float* low_oscale= (const float*)d_in[20];
  const float* mid_read  = (const float*)d_in[21];
  const float* mid_write = (const float*)d_in[22];
  const float* mid_mix   = (const float*)d_in[23];
  const float* mid_bias  = (const float*)d_in[24];
  const float* mid_oscale= (const float*)d_in[25];
  const float* high_read  = (const float*)d_in[26];
  const float* high_write = (const float*)d_in[27];
  const float* high_mix   = (const float*)d_in[28];
  const float* high_bias  = (const float*)d_in[29];
  const float* high_oscale= (const float*)d_in[30];
  const float* tg_w   = (const float*)d_in[31];
  const float* tg_b   = (const float*)d_in[32];
  const float* alpha_w= (const float*)d_in[33];
  const float* alpha_b= (const float*)d_in[34];
  const float* mem_slow_scale = (const float*)d_in[35];
  const float* mem_fast_scale = (const float*)d_in[36];
  const float* op_low_scale   = (const float*)d_in[37];
  const float* op_mid_scale   = (const float*)d_in[38];
  const float* op_high_scale  = (const float*)d_in[39];

  // ---- workspace carve-out
  char* wsb = (char*)d_ws;
  size_t off = 0;
  auto alloc = [&](size_t bytes) -> void* {
    void* p = wsb + off;
    off = (off + bytes + 255) & ~(size_t)255;
    return p;
  };
  const size_t NR = (size_t)BB * TT;           // 8192 rows
  bf16* xn_bf  = (bf16*)alloc(NR * VV * 2);
  float* x1    = (float*)alloc(NR * VV * 4);
  bf16* Wq_s = (bf16*)alloc((size_t)VV * CC * 2);
  bf16* Wk_s = (bf16*)alloc((size_t)VV * CC * 2);
  bf16* Wv_s = (bf16*)alloc((size_t)VV * CC * 2);
  bf16* Wq_f = (bf16*)alloc((size_t)VV * CC * 2);
  bf16* Wk_f = (bf16*)alloc((size_t)VV * CC * 2);
  bf16* Wv_f = (bf16*)alloc((size_t)VV * CC * 2);
  bf16* Wo_s_t = (bf16*)alloc((size_t)CC * VV * 2);
  bf16* Wo_f_t = (bf16*)alloc((size_t)CC * VV * 2);
  bf16* Wr_lo = (bf16*)alloc((size_t)VV * CC * 2);
  bf16* Wr_mi = (bf16*)alloc((size_t)VV * CC * 2);
  bf16* Wr_hi = (bf16*)alloc((size_t)VV * CC * 2);
  bf16* Ww_lo_t = (bf16*)alloc((size_t)CC * VV * 2);
  bf16* Ww_mi_t = (bf16*)alloc((size_t)CC * VV * 2);
  bf16* Ww_hi_t = (bf16*)alloc((size_t)CC * VV * 2);
  bf16* tg_w_bf    = (bf16*)alloc((size_t)CC * CC * 2);
  bf16* alpha_w_bf = (bf16*)alloc((size_t)CC * CC * 2);
  bf16* mix_lo_bf  = (bf16*)alloc((size_t)CC * CC * 2);
  bf16* mix_mi_bf  = (bf16*)alloc((size_t)CC * CC * 2);
  bf16* mix_hi_bf  = (bf16*)alloc((size_t)CC * CC * 2);
  bf16*  q_bf  = (bf16*)alloc(NR * CC * 2);   // row-major
  bf16*  kT_bf = (bf16*)alloc(NR * CC * 2);   // KS32 of k-transpose (B,C,T)
  bf16*  v_bf  = (bf16*)alloc(NR * CC * 2);   // KS32 (B,T,C)
  float* scores= (float*)alloc((size_t)TT * TT * 4);
  bf16*  sc_bf = (bf16*)alloc((size_t)TT * TT * 2);
  float* r_fast= (float*)alloc(NR * CC * 4);
  bf16*  r_s_bf= (bf16*)alloc(NR * CC * 2);
  bf16*  r_f_bf= (bf16*)alloc(NR * CC * 2);
  float* G     = (float*)alloc(NR * CC * 4);
  float* proj  = (float*)alloc(NR * VV * 4);
  bf16*  z_bf  = (bf16*)alloc(NR * CC * 2);
  float* hA    = (float*)alloc(NR * CC * 4);   // h_low f32
  float* hB    = (float*)alloc(NR * CC * 4);   // h_mid then h_high f32
  bf16*  hlo_bf= (bf16*)alloc(NR * CC * 2);
  bf16*  hmi_bf= (bf16*)alloc(NR * CC * 2);
  bf16*  hhi_bf= (bf16*)alloc(NR * CC * 2);

  const int nRC = (int)(NR * CC);
  const int nRV = (int)(NR * VV);
  const int nCC = CC * CC;
  bf16* const NOD = nullptr;
  float* const NOC = nullptr;

  // ---- 1. band weights (softmaxed q/k/v/read in KS32; plain transposed o/write in KS32)
  dim3 bw(CC);
  band_weights_kernel<<<bw,256,0,stream>>>(basis_low,  slow_q, 32, Wq_s, 1, 0);
  band_weights_kernel<<<bw,256,0,stream>>>(basis_low,  slow_k, 32, Wk_s, 1, 0);
  band_weights_kernel<<<bw,256,0,stream>>>(basis_low,  slow_v, 32, Wv_s, 1, 0);
  band_weights_kernel<<<bw,256,0,stream>>>(basis_low,  slow_o, 32, Wo_s_t, 0, 1);
  band_weights_kernel<<<bw,256,0,stream>>>(basis_high, fast_q,128, Wq_f, 1, 0);
  band_weights_kernel<<<bw,256,0,stream>>>(basis_high, fast_k,128, Wk_f, 1, 0);
  band_weights_kernel<<<bw,256,0,stream>>>(basis_high, fast_v,128, Wv_f, 1, 0);
  band_weights_kernel<<<bw,256,0,stream>>>(basis_high, fast_o,128, Wo_f_t, 0, 1);
  band_weights_kernel<<<bw,256,0,stream>>>(basis_low,  low_read,  32, Wr_lo, 1, 0);
  band_weights_kernel<<<bw,256,0,stream>>>(basis_mid,  mid_read,  64, Wr_mi, 1, 0);
  band_weights_kernel<<<bw,256,0,stream>>>(basis_high, high_read,128, Wr_hi, 1, 0);
  band_weights_kernel<<<bw,256,0,stream>>>(basis_low,  low_write, 32, Ww_lo_t, 0, 1);
  band_weights_kernel<<<bw,256,0,stream>>>(basis_mid,  mid_write, 64, Ww_mi_t, 0, 1);
  band_weights_kernel<<<bw,256,0,stream>>>(basis_high, high_write,128, Ww_hi_t, 0, 1);
  conv_ks32_kernel<<<ceil_blocks(nCC),256,0,stream>>>(tg_w,    tg_w_bf,    CC, CC, 1, 0);
  conv_ks32_kernel<<<ceil_blocks(nCC),256,0,stream>>>(alpha_w, alpha_w_bf, CC, CC, 1, 0);
  conv_ks32_kernel<<<ceil_blocks(nCC),256,0,stream>>>(low_mix,  mix_lo_bf, CC, CC, 1, 0);
  conv_ks32_kernel<<<ceil_blocks(nCC),256,0,stream>>>(mid_mix,  mix_mi_bf, CC, CC, 1, 0);
  conv_ks32_kernel<<<ceil_blocks(nCC),256,0,stream>>>(high_mix, mix_hi_bf, CC, CC, 1, 0);

  // ---- 2. phase 1: rmsnorm
  rmsnorm_bf16_kernel<<<dim3((int)NR),256,0,stream>>>(x, xn_bf, VV);

  // ---- 3. retrieval: q/k/v GEMMs write bf16 directly in the layouts the
  //         attention GEMMs need (row-major q, KS32 v, KS32-transposed k).
  const bf16* Wq[2] = {Wq_s, Wq_f};
  const bf16* Wk[2] = {Wk_s, Wk_f};
  const bf16* Wv[2] = {Wv_s, Wv_f};
  const float* dec[2] = {slow_decay, fast_decay};
  for (int p = 0; p < 2; ++p) {
    gemm_bf16_kernel<<<gemm_grid((int)NR, CC),256,0,stream>>>(xn_bf, Wq[p], NOC, q_bf, 1, (int)NR, CC, VV);
    gemm_bf16_kernel<<<gemm_grid((int)NR, CC),256,0,stream>>>(xn_bf, Wk[p], NOC, kT_bf, 3, (int)NR, CC, VV);
    gemm_bf16_kernel<<<gemm_grid((int)NR, CC),256,0,stream>>>(xn_bf, Wv[p], NOC, v_bf, 2, (int)NR, CC, VV);
    for (int b = 0; b < BB; ++b) {
      gemm_bf16_kernel<<<gemm_grid(TT, TT),256,0,stream>>>(
          q_bf + (size_t)b*TT*CC, kT_bf + (size_t)b*CC*TT, scores, NOD, 0, TT, TT, CC);
      decay_mask_kernel<<<ceil_blocks(TT*TT),256,0,stream>>>(scores, sc_bf, dec[p]);
      if (p == 0) {  // r_slow only ever consumed as bf16 A operand
        gemm_bf16_kernel<<<gemm_grid(TT, CC),256,0,stream>>>(
            sc_bf, v_bf + (size_t)b*TT*CC, NOC, r_s_bf + (size_t)b*TT*CC, 1, TT, CC, TT);
      } else {       // r_fast needs f32 for the sigmoid gate
        gemm_bf16_kernel<<<gemm_grid(TT, CC),256,0,stream>>>(
            sc_bf, v_bf + (size_t)b*TT*CC, r_fast + (size_t)b*TT*CC, NOD, 0, TT, CC, TT);
      }
    }
  }

  // ---- 4. temporal gate: tg = sigmoid(r_slow @ tg_w + tg_b); r_fast *= tg
  gemm_bf16_kernel<<<gemm_grid((int)NR, CC),256,0,stream>>>(r_s_bf, tg_w_bf, G, NOD, 0, (int)NR, CC, CC);
  gate_mul_kernel<<<ceil_blocks(nRC),256,0,stream>>>(r_fast, G, tg_b, nRC);
  conv_bf16_kernel<<<ceil_blocks(nRC),256,0,stream>>>(r_fast, r_f_bf, nRC);

  // ---- 5. x1 = x + proj_slow*slow_scale*mem_slow + proj_fast*fast_scale*mem_fast
  hipMemcpyAsync(x1, x, NR * VV * 4, hipMemcpyDeviceToDevice, stream);
  gemm_bf16_kernel<<<gemm_grid((int)NR, VV),256,0,stream>>>(r_s_bf, Wo_s_t, proj, NOD, 0, (int)NR, VV, CC);
  axpy2_kernel<<<ceil_blocks(nRV),256,0,stream>>>(x1, proj, slow_scale, mem_slow_scale, nRV);
  gemm_bf16_kernel<<<gemm_grid((int)NR, VV),256,0,stream>>>(r_f_bf, Wo_f_t, proj, NOD, 0, (int)NR, VV, CC);
  axpy2_kernel<<<ceil_blocks(nRV),256,0,stream>>>(x1, proj, fast_scale, mem_fast_scale, nRV);

  // ---- 6. phase 2: rmsnorm(x1)
  rmsnorm_bf16_kernel<<<dim3((int)NR),256,0,stream>>>(x1, xn_bf, VV);

  // low band -> hA (keep f32), hlo_bf
  gemm_bf16_kernel<<<gemm_grid((int)NR, CC),256,0,stream>>>(xn_bf, Wr_lo, NOC, z_bf, 1, (int)NR, CC, VV);
  gemm_bf16_kernel<<<gemm_grid((int)NR, CC),256,0,stream>>>(z_bf, mix_lo_bf, hA, NOD, 0, (int)NR, CC, CC);
  bias_gelu_kernel<<<ceil_blocks(nRC),256,0,stream>>>(hA, low_bias, nRC);
  conv_bf16_kernel<<<ceil_blocks(nRC),256,0,stream>>>(hA, hlo_bf, nRC);
  // mid band -> hB -> hmi_bf
  gemm_bf16_kernel<<<gemm_grid((int)NR, CC),256,0,stream>>>(xn_bf, Wr_mi, NOC, z_bf, 1, (int)NR, CC, VV);
  gemm_bf16_kernel<<<gemm_grid((int)NR, CC),256,0,stream>>>(z_bf, mix_mi_bf, hB, NOD, 0, (int)NR, CC, CC);
  bias_gelu_kernel<<<ceil_blocks(nRC),256,0,stream>>>(hB, mid_bias, nRC);
  conv_bf16_kernel<<<ceil_blocks(nRC),256,0,stream>>>(hB, hmi_bf, nRC);
  // high band -> hB (reuse f32 for gating)
  gemm_bf16_kernel<<<gemm_grid((int)NR, CC),256,0,stream>>>(xn_bf, Wr_hi, NOC, z_bf, 1, (int)NR, CC, VV);
  gemm_bf16_kernel<<<gemm_grid((int)NR, CC),256,0,stream>>>(z_bf, mix_hi_bf, hB, NOD, 0, (int)NR, CC, CC);
  bias_gelu_kernel<<<ceil_blocks(nRC),256,0,stream>>>(hB, high_bias, nRC);
  // alpha gate from h_low, applied to h_high
  gemm_bf16_kernel<<<gemm_grid((int)NR, CC),256,0,stream>>>(hlo_bf, alpha_w_bf, G, NOD, 0, (int)NR, CC, CC);
  gate_mul_kernel<<<ceil_blocks(nRC),256,0,stream>>>(hB, G, alpha_b, nRC);
  conv_bf16_kernel<<<ceil_blocks(nRC),256,0,stream>>>(hB, hhi_bf, nRC);

  // ---- 7. write-backs: out = x1 + sum of scaled band projections
  float* out = (float*)d_out;
  hipMemcpyAsync(out, x1, NR * VV * 4, hipMemcpyDeviceToDevice, stream);
  gemm_bf16_kernel<<<gemm_grid((int)NR, VV),256,0,stream>>>(hlo_bf, Ww_lo_t, proj, NOD, 0, (int)NR, VV, CC);
  axpy2_kernel<<<ceil_blocks(nRV),256,0,stream>>>(out, proj, low_oscale, op_low_scale, nRV);
  gemm_bf16_kernel<<<gemm_grid((int)NR, VV),256,0,stream>>>(hmi_bf, Ww_mi_t, proj, NOD, 0, (int)NR, VV, CC);
  axpy2_kernel<<<ceil_blocks(nRV),256,0,stream>>>(out, proj, mid_oscale, op_mid_scale, nRV);
  gemm_bf16_kernel<<<gemm_grid((int)NR, VV),256,0,stream>>>(hhi_bf, Ww_hi_t, proj, NOD, 0, (int)NR, VV, CC);
  axpy2_kernel<<<ceil_blocks(nRV),256,0,stream>>>(out, proj, high_oscale, op_high_scale, nRV);
}
